// CrystalGraphConvNet_88287347736893
// MI455X (gfx1250) — compile-verified, matrix operand order checked by reading, not run on hardware
//
#include <hip/hip_runtime.h>

typedef __bf16 bf16;
typedef __attribute__((ext_vector_type(16))) __bf16 bf16x16;
typedef __attribute__((ext_vector_type(8)))  __bf16 bf16x8;
typedef __attribute__((ext_vector_type(8)))  float  f32x8;

#define NATOM 65536
#define MNBR  12
#define ORIGF 92
#define FD    64
#define NFL   41
#define HH    128
#define NCONV 3
#define NCRY  512
#define KTOT  169      // 2*FD + NFL
#define KPAD  192      // 6 k-steps of 32
#define KSTEP 6
#define LDS_AK 200     // padded LDS row stride (halves), 400B = 16B aligned
#define EMB_LDS 104
#define NROWS (NATOM*MNBR)   // 786432
#define P1 512
#define P2 512

__device__ __forceinline__ float softplusf_(float v) {
    return v > 15.f ? v : log1pf(__expf(v));
}
__device__ __forceinline__ float sigmoidf_(float v) {
    return 1.f / (1.f + __expf(-v));
}

__device__ __forceinline__ bf16x16 frag_from_lds(const bf16* row, int kbase, int half) {
    bf16x8 lo = *(const bf16x8*)(row + kbase + half * 8);
    bf16x8 hi = *(const bf16x8*)(row + kbase + 16 + half * 8);
    return __builtin_shufflevector(lo, hi, 0,1,2,3,4,5,6,7,8,9,10,11,12,13,14,15);
}

__device__ __forceinline__ f32x8 wmma_bf16(bf16x16 a, bf16x16 b, f32x8 c) {
    return __builtin_amdgcn_wmma_f32_16x16x32_bf16(false, a, false, b, (short)0, c, false, false);
}

// ---------------- weight pre-swizzle into per-lane B-fragment layout ----------------
// B fragment (16x16x32 bf16): lane L -> col N = L&15, half = L>>4.
// elem e<8  : K = ks*32 + half*8 + e
// elem e>=8 : K = ks*32 + 16 + half*8 + (e-8)
__global__ void preswizzle_kernel(const float* __restrict__ convW,
                                  const float* __restrict__ embW,
                                  bf16* __restrict__ wfc, bf16* __restrict__ wfe) {
    int t = blockIdx.x * 256 + threadIdx.x;
    if (t < NCONV * 8 * KSTEP * 512) {
        int e = t & 15, lane = (t >> 4) & 31, ks = (t >> 9) % KSTEP;
        int ct = (t / (512 * KSTEP)) & 7, layer = t / (512 * KSTEP * 8);
        int half = lane >> 4, col = ct * 16 + (lane & 15);
        int k = ks * 32 + ((e < 8) ? (half * 8 + e) : (16 + half * 8 + (e - 8)));
        float v = (k < KTOT) ? convW[(layer * KTOT + k) * HH + col] : 0.f;
        wfc[t] = (bf16)v;
    } else {
        int u = t - NCONV * 8 * KSTEP * 512;
        if (u < 4 * 3 * 512) {
            int e = u & 15, lane = (u >> 4) & 31, ks = (u >> 9) % 3, ct = u / 1536;
            int half = lane >> 4, col = ct * 16 + (lane & 15);
            int k = ks * 32 + ((e < 8) ? (half * 8 + e) : (16 + half * 8 + (e - 8)));
            float v = (k < ORIGF) ? embW[k * FD + col] : 0.f;
            wfe[u] = (bf16)v;
        }
    }
}

// ---------------- embedding GEMM: x = af @ emb_W + b ----------------
__global__ __launch_bounds__(256) void embed_kernel(const float* __restrict__ af,
                                                    const bf16* __restrict__ wfrag,
                                                    const float* __restrict__ ebias,
                                                    float* __restrict__ x,
                                                    bf16* __restrict__ xbf) {
    __shared__ __align__(16) bf16 ldsA[32 * EMB_LDS];
    int tid = threadIdx.x;
    int r0 = blockIdx.x * 32;
    for (int i = tid; i < 32 * 96; i += 256) {
        int r = i / 96, k = i - r * 96;
        ldsA[r * EMB_LDS + k] = (k < ORIGF) ? (bf16)af[(r0 + r) * ORIGF + k] : (bf16)0.f;
    }
    __syncthreads();
    int w = tid >> 5, lane = tid & 31, lm = lane & 15, half = lane >> 4;
    int rt = w >> 2, ct = w & 3;
    f32x8 c = {0.f,0.f,0.f,0.f,0.f,0.f,0.f,0.f};
#pragma unroll
    for (int ks = 0; ks < 3; ++ks) {
        bf16x16 b = *(const bf16x16*)(wfrag + (ct * 3 + ks) * 512 + lane * 16);
        bf16x16 a = frag_from_lds(&ldsA[(rt * 16 + lm) * EMB_LDS], ks * 32, half);
        c = wmma_bf16(a, b, c);
    }
    int col = ct * 16 + lm;
    float bc = ebias[col];
#pragma unroll
    for (int e = 0; e < 8; ++e) {
        int row = r0 + rt * 16 + half * 8 + e;
        float v = c[e] + bc;
        x[row * FD + col] = v;
        xbf[row * FD + col] = (bf16)v;
    }
}

// ---------------- conv pass A: GEMM + BN1 stats (no z store) ----------------
__global__ __launch_bounds__(256) void conv_pass_a(const bf16* __restrict__ xbf,
                                                   const float* __restrict__ nf,
                                                   const int* __restrict__ idx,
                                                   const bf16* __restrict__ wfrag,
                                                   const float* __restrict__ bias,
                                                   float* __restrict__ st1) {
    __shared__ __align__(16) bf16 ldsA[48 * LDS_AK];
    __shared__ int ldsIdx[48];
    const int tid = threadIdx.x;
    const int n0 = blockIdx.x * 4;
    if (tid < 48) ldsIdx[tid] = idx[n0 * MNBR + tid];
    __syncthreads();
    for (int i = tid; i < 48 * KPAD; i += 256) {
        int r = i / KPAD, k = i - r * KPAD;
        int n = n0 + r / MNBR, m = r - (r / MNBR) * MNBR;
        bf16 v;
        if (k < FD)             v = xbf[n * FD + k];
        else if (k < 2 * FD)    v = xbf[ldsIdx[r] * FD + (k - FD)];
        else if (k < KTOT)      v = (bf16)nf[(n * MNBR + m) * NFL + (k - 2 * FD)];
        else                    v = (bf16)0.f;
        ldsA[r * LDS_AK + k] = v;
    }
    __syncthreads();
    int w = tid >> 5, lane = tid & 31, lm = lane & 15, half = lane >> 4;
    f32x8 cc0 = {0.f,0.f,0.f,0.f,0.f,0.f,0.f,0.f};
    f32x8 cc1 = cc0, cc2 = cc0;
    const bf16* wf = wfrag + (w * KSTEP) * 512;
#pragma unroll
    for (int ks = 0; ks < KSTEP; ++ks) {
        bf16x16 b = *(const bf16x16*)(wf + ks * 512 + lane * 16);
        int kbase = ks * 32;
        bf16x16 a0 = frag_from_lds(&ldsA[(0 * 16 + lm) * LDS_AK], kbase, half);
        bf16x16 a1 = frag_from_lds(&ldsA[(1 * 16 + lm) * LDS_AK], kbase, half);
        bf16x16 a2 = frag_from_lds(&ldsA[(2 * 16 + lm) * LDS_AK], kbase, half);
        cc0 = wmma_bf16(a0, b, cc0);
        cc1 = wmma_bf16(a1, b, cc1);
        cc2 = wmma_bf16(a2, b, cc2);
    }
    int col = w * 16 + lm;
    float bcol = bias[col];
    float s = 0.f, q = 0.f;
    f32x8 cc[3] = {cc0, cc1, cc2};
#pragma unroll
    for (int t = 0; t < 3; ++t)
#pragma unroll
        for (int e = 0; e < 8; ++e) { float z = cc[t][e] + bcol; s += z; q += z * z; }
    s += __shfl_xor(s, 16, 32);
    q += __shfl_xor(q, 16, 32);
    if (lane < 16) {
        int slot = blockIdx.x & (P1 - 1);
        atomicAdd(&st1[slot * 256 + col], s);
        atomicAdd(&st1[slot * 256 + 128 + col], q);
    }
}

// ---------------- conv pass B: recompute GEMM, BN1 apply, gate+sum over M, BN2 stats ----------------
__global__ __launch_bounds__(256) void conv_pass_b(const bf16* __restrict__ xbf,
                                                   const float* __restrict__ nf,
                                                   const int* __restrict__ idx,
                                                   const bf16* __restrict__ wfrag,
                                                   const float* __restrict__ bias,
                                                   const float* __restrict__ coef1,
                                                   float* __restrict__ xsum,
                                                   float* __restrict__ st2) {
    __shared__ __align__(16) bf16 ldsA[48 * LDS_AK];
    __shared__ int ldsIdx[48];
    __shared__ float ldsZ[48 * 132];
    const int tid = threadIdx.x;
    const int n0 = blockIdx.x * 4;
    if (tid < 48) ldsIdx[tid] = idx[n0 * MNBR + tid];
    __syncthreads();
    for (int i = tid; i < 48 * KPAD; i += 256) {
        int r = i / KPAD, k = i - r * KPAD;
        int n = n0 + r / MNBR, m = r - (r / MNBR) * MNBR;
        bf16 v;
        if (k < FD)             v = xbf[n * FD + k];
        else if (k < 2 * FD)    v = xbf[ldsIdx[r] * FD + (k - FD)];
        else if (k < KTOT)      v = (bf16)nf[(n * MNBR + m) * NFL + (k - 2 * FD)];
        else                    v = (bf16)0.f;
        ldsA[r * LDS_AK + k] = v;
    }
    __syncthreads();
    int w = tid >> 5, lane = tid & 31, lm = lane & 15, half = lane >> 4;
    f32x8 cc0 = {0.f,0.f,0.f,0.f,0.f,0.f,0.f,0.f};
    f32x8 cc1 = cc0, cc2 = cc0;
    const bf16* wf = wfrag + (w * KSTEP) * 512;
#pragma unroll
    for (int ks = 0; ks < KSTEP; ++ks) {
        bf16x16 b = *(const bf16x16*)(wf + ks * 512 + lane * 16);
        int kbase = ks * 32;
        bf16x16 a0 = frag_from_lds(&ldsA[(0 * 16 + lm) * LDS_AK], kbase, half);
        bf16x16 a1 = frag_from_lds(&ldsA[(1 * 16 + lm) * LDS_AK], kbase, half);
        bf16x16 a2 = frag_from_lds(&ldsA[(2 * 16 + lm) * LDS_AK], kbase, half);
        cc0 = wmma_bf16(a0, b, cc0);
        cc1 = wmma_bf16(a1, b, cc1);
        cc2 = wmma_bf16(a2, b, cc2);
    }
    int col = w * 16 + lm;
    float bcol = bias[col];
    float scale = coef1[col], shift = coef1[128 + col];
    f32x8 cc[3] = {cc0, cc1, cc2};
#pragma unroll
    for (int t = 0; t < 3; ++t)
#pragma unroll
        for (int e = 0; e < 8; ++e) {
            int row = t * 16 + half * 8 + e;
            ldsZ[row * 132 + col] = (cc[t][e] + bcol) * scale + shift;
        }
    __syncthreads();
    int nl = tid >> 6, k = tid & 63;
    float acc = 0.f;
#pragma unroll
    for (int m = 0; m < MNBR; ++m) {
        float f = ldsZ[(nl * MNBR + m) * 132 + k];
        float g = ldsZ[(nl * MNBR + m) * 132 + FD + k];
        acc += sigmoidf_(f) * softplusf_(g);
    }
    xsum[(n0 + nl) * FD + k] = acc;
    int slot = blockIdx.x & (P2 - 1);
    atomicAdd(&st2[slot * 128 + k], acc);
    atomicAdd(&st2[slot * 128 + 64 + k], acc * acc);
}

// ---------------- BN stat folds ----------------
__global__ void reduce_bn1(const float* __restrict__ st1, const float* __restrict__ g,
                           const float* __restrict__ b, float* __restrict__ coef1) {
    int c = threadIdx.x; // 128
    float S = 0.f, Q = 0.f;
    for (int s = 0; s < P1; ++s) { S += st1[s * 256 + c]; Q += st1[s * 256 + 128 + c]; }
    const float R = (float)NROWS;
    float mu = S / R;
    float var = Q / R - mu * mu;
    float rs = rsqrtf(var + 1e-5f);
    float sc = g[c] * rs;
    coef1[c] = sc;
    coef1[128 + c] = b[c] - mu * sc;
}

__global__ void reduce_bn2(const float* __restrict__ st2, const float* __restrict__ g,
                           const float* __restrict__ b, float* __restrict__ coef2) {
    int c = threadIdx.x; // 64
    float S = 0.f, Q = 0.f;
    for (int s = 0; s < P2; ++s) { S += st2[s * 128 + c]; Q += st2[s * 128 + 64 + c]; }
    const float R = (float)NATOM;
    float mu = S / R;
    float var = Q / R - mu * mu;
    float rs = rsqrtf(var + 1e-5f);
    float sc = g[c] * rs;
    coef2[c] = sc;
    coef2[64 + c] = b[c] - mu * sc;
}

// ---------------- conv pass C: residual softplus, refresh bf16 copy ----------------
__global__ void conv_pass_c(float* __restrict__ x, bf16* __restrict__ xbf,
                            const float* __restrict__ xsum, const float* __restrict__ coef2) {
    int i = blockIdx.x * 256 + threadIdx.x;
    int k = i & 63;
    float s = xsum[i] * coef2[k] + coef2[64 + k];
    float v = softplusf_(x[i] + s);
    x[i] = v;
    xbf[i] = (bf16)v;
}

// ---------------- pooling + head ----------------
__global__ void zero_kernel(float* __restrict__ p, int n) {
    int i = blockIdx.x * 256 + threadIdx.x;
    if (i < n) p[i] = 0.f;
}

__global__ void pool_kernel(const float* __restrict__ x, const int* __restrict__ cry,
                            float* __restrict__ psum, float* __restrict__ pcnt) {
    int i = blockIdx.x * 256 + threadIdx.x; // N*FD
    int n = i >> 6, k = i & 63;
    int c = cry[n];
    atomicAdd(&psum[c * FD + k], x[i]);
    if (k == 0) atomicAdd(&pcnt[c], 1.f);
}

__global__ void head_pool_act(const float* __restrict__ psum, const float* __restrict__ pcnt,
                              float* __restrict__ crys) {
    int i = blockIdx.x * 256 + threadIdx.x; // NCRY*FD
    int c = i >> 6;
    float cnt = fmaxf(pcnt[c], 1.f);
    crys[i] = softplusf_(psum[i] / cnt);
}

__global__ void head_gemm(const float* __restrict__ in, const float* __restrict__ W,
                          const float* __restrict__ b, float* __restrict__ out,
                          int K, int act) {
    int i = blockIdx.x * 256 + threadIdx.x; // NCRY*HH
    int r = i >> 7, c = i & 127;
    float acc = b[c];
    for (int k = 0; k < K; ++k) acc += in[r * K + k] * W[k * HH + c];
    out[i] = act ? softplusf_(acc) : acc;
}

// ---------------- host launch ----------------
extern "C" void kernel_launch(void* const* d_in, const int* in_sizes, int n_in,
                              void* d_out, int out_size, void* d_ws, size_t ws_size,
                              hipStream_t stream) {
    const float* af   = (const float*)d_in[0];
    const float* nf   = (const float*)d_in[1];
    const int*   idx  = (const int*)d_in[2];
    const int*   cry  = (const int*)d_in[3];
    const float* embW = (const float*)d_in[5];
    const float* embB = (const float*)d_in[6];
    const float* convW= (const float*)d_in[7];
    const float* convB= (const float*)d_in[8];
    const float* bn1g = (const float*)d_in[9];
    const float* bn1b = (const float*)d_in[10];
    const float* bn2g = (const float*)d_in[11];
    const float* bn2b = (const float*)d_in[12];
    const float* fcW  = (const float*)d_in[13];
    const float* fcB  = (const float*)d_in[14];
    const float* h1W  = (const float*)d_in[15];
    const float* h1B  = (const float*)d_in[16];
    const float* h2W  = (const float*)d_in[17];
    const float* h2B  = (const float*)d_in[18];

    char* p = (char*)d_ws;
    auto carve = [&](size_t bytes) -> void* {
        void* r = (void*)p;
        p += (bytes + 255) & ~(size_t)255;
        return r;
    };
    float* x     = (float*)carve((size_t)NATOM * FD * 4);
    bf16*  xbf   = (bf16*) carve((size_t)NATOM * FD * 2);
    float* xsum  = (float*)carve((size_t)NATOM * FD * 4);
    bf16*  wfc   = (bf16*) carve((size_t)NCONV * 8 * KSTEP * 512 * 2);
    bf16*  wfe   = (bf16*) carve((size_t)4 * 3 * 512 * 2);
    float* st1   = (float*)carve((size_t)P1 * 256 * 4);
    float* st2   = (float*)carve((size_t)P2 * 128 * 4);
    float* coef1 = (float*)carve(256 * 4);
    float* coef2 = (float*)carve(128 * 4);
    float* psum  = (float*)carve((size_t)NCRY * FD * 4);
    float* pcnt  = (float*)carve((size_t)NCRY * 4);
    float* crys  = (float*)carve((size_t)NCRY * FD * 4);
    float* t1    = (float*)carve((size_t)NCRY * HH * 4);
    float* t2    = (float*)carve((size_t)NCRY * HH * 4);

    // 1. swizzle weights into WMMA B-fragment layout (bf16)
    preswizzle_kernel<<<312, 256, 0, stream>>>(convW, embW, wfc, wfe);

    // 2. embedding GEMM
    embed_kernel<<<NATOM / 32, 256, 0, stream>>>(af, wfe, embB, x, xbf);

    // 3. conv layers (two-pass: stats, then recompute+apply)
    for (int L = 0; L < NCONV; ++L) {
        zero_kernel<<<(P1 * 256 + 255) / 256, 256, 0, stream>>>(st1, P1 * 256);
        conv_pass_a<<<NATOM / 4, 256, 0, stream>>>(xbf, nf, idx, wfc + (size_t)L * 8 * KSTEP * 512,
                                                   convB + L * 128, st1);
        reduce_bn1<<<1, 128, 0, stream>>>(st1, bn1g + L * 128, bn1b + L * 128, coef1);
        zero_kernel<<<(P2 * 128 + 255) / 256, 256, 0, stream>>>(st2, P2 * 128);
        conv_pass_b<<<NATOM / 4, 256, 0, stream>>>(xbf, nf, idx, wfc + (size_t)L * 8 * KSTEP * 512,
                                                   convB + L * 128, coef1, xsum, st2);
        reduce_bn2<<<1, 64, 0, stream>>>(st2, bn2g + L * 64, bn2b + L * 64, coef2);
        conv_pass_c<<<NATOM * FD / 256, 256, 0, stream>>>(x, xbf, xsum, coef2);
    }

    // 4. crystal mean pooling
    zero_kernel<<<(NCRY * FD + 255) / 256, 256, 0, stream>>>(psum, NCRY * FD);
    zero_kernel<<<(NCRY + 255) / 256, 256, 0, stream>>>(pcnt, NCRY);
    pool_kernel<<<NATOM * FD / 256, 256, 0, stream>>>(x, cry, psum, pcnt);
    head_pool_act<<<(NCRY * FD + 255) / 256, 256, 0, stream>>>(psum, pcnt, crys);

    // 5. head MLP (tiny: 512 rows)
    head_gemm<<<NCRY * HH / 256, 256, 0, stream>>>(crys, fcW, fcB, t1, FD, 1);
    head_gemm<<<NCRY * HH / 256, 256, 0, stream>>>(t1, h1W, h1B, t2, HH, 1);
    head_gemm<<<NCRY * HH / 256, 256, 0, stream>>>(t2, h2W, h2B, (float*)d_out, HH, 0);
}
